// GNNNeighborPred_2181843386577
// MI455X (gfx1250) — matrix operand order repaired and checked by reading, not compiled, not run on hardware
//
#include <hip/hip_runtime.h>

// ---------------- problem constants (match reference) ----------------
constexpr int NODE_NUM = 200000;
constexpr int H  = 768;
constexpr int NH = 8;
constexpr int DH = 96;
constexpr int Bc = 2048;
constexpr int Kc = 8;
constexpr int Fk = 5;          // neighbors per relation
constexpr int Rr = 2;          // relations
constexpr int Ll = 2;          // layers
constexpr int NSEED = Bc * Kc; // 16384 seed nodes
constexpr int NTW = 6;         // N-tiles per wave (48 tiles / 8 waves)

using bf16 = __bf16;
typedef __attribute__((ext_vector_type(16))) bf16  v16bf;
typedef __attribute__((ext_vector_type(8)))  bf16  v8bf;
typedef __attribute__((ext_vector_type(8)))  float v8f;

// ---------------- helpers ----------------
// native f32 -> bf16 (lowers to v_cvt_pk_bf16_f32 when pairs are adjacent)
__device__ __forceinline__ bf16 f2bf(float f) { return (bf16)f; }

// A fragment: lane holds row M = lane&15; K chunks [k0+hs*8, +8) and [k0+16+hs*8, +8)
__device__ __forceinline__ v16bf load_a_bf(const bf16* __restrict__ row, int k0, int hs) {
    v8bf lo = *reinterpret_cast<const v8bf*>(row + k0 + hs * 8);
    v8bf hi = *reinterpret_cast<const v8bf*>(row + k0 + 16 + hs * 8);
    v16bf a;
#pragma unroll
    for (int e = 0; e < 8; ++e) { a[e] = lo[e]; a[8 + e] = hi[e]; }
    return a;
}

__device__ __forceinline__ v16bf load_a_f32(const float* __restrict__ row, int k0, int hs) {
    const float4* p0 = reinterpret_cast<const float4*>(row + k0 + hs * 8);
    const float4* p1 = reinterpret_cast<const float4*>(row + k0 + 16 + hs * 8);
    float4 x0 = p0[0], x1 = p0[1], y0 = p1[0], y1 = p1[1];
    v16bf a;
    a[0] = f2bf(x0.x); a[1]  = f2bf(x0.y); a[2]  = f2bf(x0.z); a[3]  = f2bf(x0.w);
    a[4] = f2bf(x1.x); a[5]  = f2bf(x1.y); a[6]  = f2bf(x1.z); a[7]  = f2bf(x1.w);
    a[8] = f2bf(y0.x); a[9]  = f2bf(y0.y); a[10] = f2bf(y0.z); a[11] = f2bf(y0.w);
    a[12] = f2bf(y1.x); a[13] = f2bf(y1.y); a[14] = f2bf(y1.z); a[15] = f2bf(y1.w);
    return a;
}

// B fragment: weights stored transposed (wT[n][k]); lane holds col N = lane&15,
// K chunks [k0+hs*8, +8) and [k0+16+hs*8, +8)  (16B aligned loads)
__device__ __forceinline__ v16bf load_b(const bf16* __restrict__ wT, int col, int k0, int hs) {
    const bf16* p = wT + (size_t)col * H + k0 + hs * 8;
    v8bf lo = *reinterpret_cast<const v8bf*>(p);
    v8bf hi = *reinterpret_cast<const v8bf*>(p + 16);
    v16bf b;
#pragma unroll
    for (int e = 0; e < 8; ++e) { b[e] = lo[e]; b[8 + e] = hi[e]; }
    return b;
}

__device__ __forceinline__ v8f wmma_bf16(v16bf a, v16bf b, v8f c) {
    return __builtin_amdgcn_wmma_f32_16x16x32_bf16(false, a, false, b, (short)0, c, false, false);
}

// ---------------- prep kernels ----------------
// dst[m][n][k] = (bf16) src[m][k][n]   (nmat matrices of HxH)
__global__ void k_transpose_bf(const float* __restrict__ src, bf16* __restrict__ dst, int total) {
    for (int i = blockIdx.x * blockDim.x + threadIdx.x; i < total; i += gridDim.x * blockDim.x) {
        int m = i / (H * H);
        int rem = i - m * (H * H);
        int k = rem / H;
        int n = rem - k * H;
        dst[(size_t)m * H * H + (size_t)n * H + k] = f2bf(src[i]);
    }
}

__global__ void k_convert_bf(const float* __restrict__ src, bf16* __restrict__ dst, int total) {
    for (int i = blockIdx.x * blockDim.x + threadIdx.x; i < total; i += gridDim.x * blockDim.x)
        dst[i] = f2bf(src[i]);
}

// h[n][:] = emb[id_pairs_flat[n]][:]  (f32 + bf16 copies)
__global__ void k_gather_h(const float* __restrict__ emb, const int* __restrict__ ids,
                           float* __restrict__ h, bf16* __restrict__ h_bf, int total) {
    for (int i = blockIdx.x * blockDim.x + threadIdx.x; i < total; i += gridDim.x * blockDim.x) {
        int n = i / H;
        int c = i - n * H;
        float v = emb[(size_t)ids[n] * H + c];
        h[i] = v;
        h_bf[i] = f2bf(v);
    }
}

// ---------------- fused attention kernel ----------------
// One workgroup (256 threads = 8 waves) per 16-seed-node tile.
// Loops over relations; writes agg[16][768] for its tile.
// Each wave owns a contiguous group of NTW=6 N-tiles; K is the OUTER gemm loop
// so each K-step does 1 A-frag load + 6 B-frag loads + 6 independent WMMAs.
__global__ void __launch_bounds__(256)
k_attention(const float* __restrict__ emb,
            const bf16* __restrict__ h_bf,
            float* __restrict__ agg,
            const bf16* __restrict__ wqT,   // [R][H][H] transposed bf16 (layer slice)
            const bf16* __restrict__ wkT,
            const bf16* __restrict__ wvT,
            const int* __restrict__ src_ids,        // [R][N][F]
            const unsigned char* __restrict__ mask) // [R][N][F]
{
    __shared__ float buf[16][H];       // q, then agg accumulator (48 KB)
    __shared__ float sc[16][Fk][NH];   // scores -> alpha (2.5 KB)
    __shared__ int   srcid[16 * Fk];   // gathered ids for this tile

    const int tid  = threadIdx.x;
    const int lane = tid & 31;
    const int wave = tid >> 5;
    const int hs   = lane >> 4;
    const int nn   = lane & 15;
    const int ntBase = wave * NTW;     // this wave's N-tile group
    const int nodeBase = blockIdx.x * 16;
    const float inv_sqrt_dh = 0.10206207261596577f; // 1/sqrt(96)

    for (int r = 0; r < Rr; ++r) {
        // gather src ids (mask -> PAD row) and zero scores
        if (tid < 16 * Fk) {
            int row = tid;
            int n = nodeBase + row / Fk;
            int f = row - (row / Fk) * Fk;
            size_t idx = ((size_t)r * NSEED + n) * Fk + f;
            int sid = src_ids[idx];
            if (mask[idx]) sid = NODE_NUM;
            srcid[row] = sid;
        }
        for (int i = tid; i < 16 * Fk * NH; i += 256) ((float*)sc)[i] = 0.0f;
        __syncthreads();

        // ---- Phase A: q = h_tile @ Wq  -> buf ----
        {
            const bf16* wq = wqT + (size_t)r * H * H;
            const bf16* arow = h_bf + (size_t)(nodeBase + nn) * H;
            v8f c[NTW] = {};
            for (int k0 = 0; k0 < H; k0 += 32) {
                v16bf a = load_a_bf(arow, k0, hs);
#pragma unroll
                for (int j = 0; j < NTW; ++j) {
                    v16bf b = load_b(wq, (ntBase + j) * 16 + nn, k0, hs);
                    c[j] = wmma_bf16(a, b, c[j]);
                }
            }
#pragma unroll
            for (int j = 0; j < NTW; ++j)
#pragma unroll
                for (int v = 0; v < 8; ++v)
                    buf[v + 8 * hs][(ntBase + j) * 16 + nn] = c[j][v];
        }
        __syncthreads();

        // ---- Phase B: kk = sf @ Wk, scatter q.kk into scores ----
        {
            const bf16* wk = wkT + (size_t)r * H * H;
            for (int m = 0; m < 5; ++m) {    // 5 M-tiles cover 80 = 16*F rows
                const float* arow = emb + (size_t)srcid[m * 16 + nn] * H;
                v8f c[NTW] = {};
                for (int k0 = 0; k0 < H; k0 += 32) {
                    v16bf a = load_a_f32(arow, k0, hs);
#pragma unroll
                    for (int j = 0; j < NTW; ++j) {
                        v16bf b = load_b(wk, (ntBase + j) * 16 + nn, k0, hs);
                        c[j] = wmma_bf16(a, b, c[j]);
                    }
                }
#pragma unroll
                for (int j = 0; j < NTW; ++j) {
                    int col  = (ntBase + j) * 16 + nn;
                    int head = col / DH;
#pragma unroll
                    for (int v = 0; v < 8; ++v) {
                        int row = m * 16 + v + 8 * hs;
                        int nl  = row / Fk;
                        int f   = row - nl * Fk;
                        atomicAdd(&sc[nl][f][head], c[j][v] * buf[nl][col]);
                    }
                }
            }
        }
        __syncthreads();

        // ---- Phase C: softmax over F per (node, head) ----
        if (tid < 16 * NH) {
            int nl = tid >> 3, hd = tid & 7;
            float s[Fk], mx = -3.0e38f;
#pragma unroll
            for (int f = 0; f < Fk; ++f) { s[f] = sc[nl][f][hd] * inv_sqrt_dh; mx = fmaxf(mx, s[f]); }
            float sum = 0.0f;
#pragma unroll
            for (int f = 0; f < Fk; ++f) { s[f] = __expf(s[f] - mx); sum += s[f]; }
            float inv = 1.0f / sum;
#pragma unroll
            for (int f = 0; f < Fk; ++f) sc[nl][f][hd] = s[f] * inv;
        }
        __syncthreads();
        for (int i = tid; i < 16 * H; i += 256) ((float*)buf)[i] = 0.0f;
        __syncthreads();

        // ---- Phase D: vv = sf @ Wv, accumulate alpha * vv into buf ----
        {
            const bf16* wv = wvT + (size_t)r * H * H;
            for (int m = 0; m < 5; ++m) {
                const float* arow = emb + (size_t)srcid[m * 16 + nn] * H;
                v8f c[NTW] = {};
                for (int k0 = 0; k0 < H; k0 += 32) {
                    v16bf a = load_a_f32(arow, k0, hs);
#pragma unroll
                    for (int j = 0; j < NTW; ++j) {
                        v16bf b = load_b(wv, (ntBase + j) * 16 + nn, k0, hs);
                        c[j] = wmma_bf16(a, b, c[j]);
                    }
                }
#pragma unroll
                for (int j = 0; j < NTW; ++j) {
                    int col  = (ntBase + j) * 16 + nn;
                    int head = col / DH;
#pragma unroll
                    for (int v = 0; v < 8; ++v) {
                        int row = m * 16 + v + 8 * hs;
                        int nl  = row / Fk;
                        int f   = row - nl * Fk;
                        atomicAdd(&buf[nl][col], sc[nl][f][head] * c[j][v]);
                    }
                }
            }
        }
        __syncthreads();

        // ---- write / accumulate tile agg ----
        float* aggRow = agg + (size_t)nodeBase * H;
        for (int i = tid; i < 16 * H; i += 256) {
            float v = ((float*)buf)[i];
            if (r == 0) aggRow[i] = v;
            else        aggRow[i] += v;
        }
        __syncthreads();
    }
}

// ---------------- o = agg @ Wo, h = LN(h + o) ----------------
__global__ void __launch_bounds__(256)
k_out_ln(const float* __restrict__ agg, float* __restrict__ h, bf16* __restrict__ h_bf,
         const bf16* __restrict__ woT, const float* __restrict__ g, const float* __restrict__ bia)
{
    __shared__ float buf[16][H];
    const int tid  = threadIdx.x;
    const int lane = tid & 31;
    const int wave = tid >> 5;
    const int hs   = lane >> 4;
    const int nn   = lane & 15;
    const int ntBase = wave * NTW;
    const int nodeBase = blockIdx.x * 16;

    // GEMM: o_tile = agg_tile @ Wo  (K outer, 6 accumulators)
    {
        const float* arow = agg + (size_t)(nodeBase + nn) * H;
        v8f c[NTW] = {};
        for (int k0 = 0; k0 < H; k0 += 32) {
            v16bf a = load_a_f32(arow, k0, hs);
#pragma unroll
            for (int j = 0; j < NTW; ++j) {
                v16bf b = load_b(woT, (ntBase + j) * 16 + nn, k0, hs);
                c[j] = wmma_bf16(a, b, c[j]);
            }
        }
#pragma unroll
        for (int j = 0; j < NTW; ++j)
#pragma unroll
            for (int v = 0; v < 8; ++v)
                buf[v + 8 * hs][(ntBase + j) * 16 + nn] = c[j][v];
    }
    __syncthreads();

    // residual
    for (int i = tid; i < 16 * H; i += 256)
        ((float*)buf)[i] += h[(size_t)nodeBase * H + i];
    __syncthreads();

    // per-row LayerNorm (each wave handles 2 rows)
    for (int rr = wave; rr < 16; rr += 8) {
        float s = 0.0f, ss = 0.0f;
        for (int c2 = lane; c2 < H; c2 += 32) {
            float x = buf[rr][c2];
            s += x; ss += x * x;
        }
#pragma unroll
        for (int off = 16; off > 0; off >>= 1) {
            s  += __shfl_xor(s, off, 32);
            ss += __shfl_xor(ss, off, 32);
        }
        float mean = s * (1.0f / H);
        float var  = ss * (1.0f / H) - mean * mean;
        float rstd = rsqrtf(var + 1e-5f);
        for (int c2 = lane; c2 < H; c2 += 32) {
            float y = (buf[rr][c2] - mean) * rstd * g[c2] + bia[c2];
            size_t idx = (size_t)(nodeBase + rr) * H + c2;
            h[idx] = y;
            h_bf[idx] = f2bf(y);
        }
    }
}

// ---------------- word = aver_feats @ Wt.T ----------------
__global__ void __launch_bounds__(256)
k_word(const bf16* __restrict__ aver_bf, const bf16* __restrict__ wtT, float* __restrict__ word)
{
    const int lane = threadIdx.x & 31;
    const int wave = threadIdx.x >> 5;
    const int hs   = lane >> 4;
    const int nn   = lane & 15;
    const int ntBase = wave * NTW;
    const int base = blockIdx.x * 16;

    const bf16* arow = aver_bf + (size_t)(base + nn) * H;
    v8f c[NTW] = {};
    for (int k0 = 0; k0 < H; k0 += 32) {
        v16bf a = load_a_bf(arow, k0, hs);
#pragma unroll
        for (int j = 0; j < NTW; ++j) {
            v16bf b = load_b(wtT, (ntBase + j) * 16 + nn, k0, hs);
            c[j] = wmma_bf16(a, b, c[j]);
        }
    }
#pragma unroll
    for (int j = 0; j < NTW; ++j)
#pragma unroll
        for (int v = 0; v < 8; ++v)
            word[(size_t)(base + v + 8 * hs) * H + (ntBase + j) * 16 + nn] = c[j][v];
}

// ---------------- out[b][k] = dot(h[b*K+k], word[b]) ----------------
__global__ void __launch_bounds__(256)
k_final(const float* __restrict__ h, const float* __restrict__ word, float* __restrict__ out)
{
    const int b    = blockIdx.x;
    const int lane = threadIdx.x & 31;
    const int w    = threadIdx.x >> 5; // k index, 0..7
    const int n    = b * Kc + w;
    const float* hr = h + (size_t)n * H;
    const float* wr = word + (size_t)b * H;
    float p = 0.0f;
    for (int c = lane; c < H; c += 32)
        p = fmaf(hr[c], wr[c], p);
#pragma unroll
    for (int off = 16; off > 0; off >>= 1)
        p += __shfl_xor(p, off, 32);
    if (lane == 0) out[n] = p;
}

// ---------------- host ----------------
extern "C" void kernel_launch(void* const* d_in, const int* in_sizes, int n_in,
                              void* d_out, int out_size, void* d_ws, size_t ws_size,
                              hipStream_t stream) {
    (void)in_sizes; (void)n_in; (void)out_size; (void)ws_size;

    const float* emb      = (const float*)d_in[0];
    const float* aver     = (const float*)d_in[1];
    const float* Wt       = (const float*)d_in[2];
    const float* Wq       = (const float*)d_in[3];
    const float* Wk       = (const float*)d_in[4];
    const float* Wv       = (const float*)d_in[5];
    const float* Wo       = (const float*)d_in[6];
    const float* ln_g     = (const float*)d_in[7];
    const float* ln_b     = (const float*)d_in[8];
    const int*   id_pairs = (const int*)d_in[9];
    const int*   src_ids  = (const int*)d_in[10];
    const unsigned char* mask = (const unsigned char*)d_in[11];
    float* out = (float*)d_out;

    // workspace carve-up
    char* ws = (char*)d_ws;
    size_t off = 0;
    auto take = [&](size_t bytes) -> char* {
        char* p = ws + off;
        off = (off + bytes + 255) & ~(size_t)255;
        return p;
    };
    bf16*  wq_bf   = (bf16*)take((size_t)Ll * Rr * H * H * sizeof(bf16));
    bf16*  wk_bf   = (bf16*)take((size_t)Ll * Rr * H * H * sizeof(bf16));
    bf16*  wv_bf   = (bf16*)take((size_t)Ll * Rr * H * H * sizeof(bf16));
    bf16*  wo_bf   = (bf16*)take((size_t)Ll * H * H * sizeof(bf16));
    bf16*  wt_bf   = (bf16*)take((size_t)H * H * sizeof(bf16));
    bf16*  aver_bf = (bf16*)take((size_t)Bc * H * sizeof(bf16));
    float* h       = (float*)take((size_t)NSEED * H * sizeof(float));
    bf16*  h_bf    = (bf16*)take((size_t)NSEED * H * sizeof(bf16));
    float* agg     = (float*)take((size_t)NSEED * H * sizeof(float));
    float* word    = (float*)take((size_t)Bc * H * sizeof(float));

    auto blocks = [](int total) { return (total + 255) / 256; };

    // weight prep (bf16, B-operands pre-transposed for the WMMA B layout)
    int tqkv = Ll * Rr * H * H;
    k_transpose_bf<<<blocks(tqkv), 256, 0, stream>>>(Wq, wq_bf, tqkv);
    k_transpose_bf<<<blocks(tqkv), 256, 0, stream>>>(Wk, wk_bf, tqkv);
    k_transpose_bf<<<blocks(tqkv), 256, 0, stream>>>(Wv, wv_bf, tqkv);
    int to = Ll * H * H;
    k_transpose_bf<<<blocks(to), 256, 0, stream>>>(Wo, wo_bf, to);
    // word GEMM B = Wt.T; its transpose is Wt itself -> plain convert
    k_convert_bf<<<blocks(H * H), 256, 0, stream>>>(Wt, wt_bf, H * H);
    k_convert_bf<<<blocks(Bc * H), 256, 0, stream>>>(aver, aver_bf, Bc * H);

    // initial hidden state gather
    k_gather_h<<<blocks(NSEED * H), 256, 0, stream>>>(emb, id_pairs, h, h_bf, NSEED * H);

    // layers
    for (int l = 0; l < Ll; ++l) {
        k_attention<<<NSEED / 16, 256, 0, stream>>>(
            emb, h_bf, agg,
            wq_bf + (size_t)l * Rr * H * H,
            wk_bf + (size_t)l * Rr * H * H,
            wv_bf + (size_t)l * Rr * H * H,
            src_ids, mask);
        k_out_ln<<<NSEED / 16, 256, 0, stream>>>(
            agg, h, h_bf, wo_bf + (size_t)l * H * H, ln_g + (size_t)l * H, ln_b + (size_t)l * H);
    }

    // prediction head
    k_word<<<Bc / 16, 256, 0, stream>>>(aver_bf, wt_bf, word);
    k_final<<<Bc, 256, 0, stream>>>(h, word, out);
}
